// AttentionBlock_34222299414888
// MI455X (gfx1250) — matrix operand order repaired and checked by reading, not compiled, not run on hardware
//
#include <hip/hip_runtime.h>
#include <hip/hip_bf16.h>

typedef __attribute__((ext_vector_type(16))) _Float16 v16h;
typedef __attribute__((ext_vector_type(8)))  _Float16 v8h;
typedef __attribute__((ext_vector_type(8)))  float    v8f;

#define SHUF16(lo, hi) __builtin_shufflevector((lo), (hi), 0,1,2,3,4,5,6,7,8,9,10,11,12,13,14,15)

// ---------------------------------------------------------------------------
// K0: convert f32 weights to f16
// ---------------------------------------------------------------------------
__global__ __launch_bounds__(256) void cvt_weights(const float* __restrict__ qw,
                                                   const float* __restrict__ pw,
                                                   _Float16* __restrict__ qwh,
                                                   _Float16* __restrict__ pwh,
                                                   int nq, int np) {
  int i = blockIdx.x * 256 + threadIdx.x;
  if (i < nq) qwh[i] = (_Float16)qw[i];
  if (i < np) pwh[i] = (_Float16)pw[i];
}

// ---------------------------------------------------------------------------
// K1: GroupNorm, writes xnT[b][p][c] (f16), one block per (batch, group)
// ---------------------------------------------------------------------------
__global__ __launch_bounds__(256) void groupnorm_kernel(const float* __restrict__ x,
                                                        const float* __restrict__ gw,
                                                        const float* __restrict__ gb,
                                                        _Float16* __restrict__ xnT,
                                                        int C, int HW, int CG) {
  int blk = blockIdx.x;
  int b = blk >> 3, g = blk & 7;
  int c0 = g * CG;
  const float* xb = x + ((size_t)b * C + c0) * HW;
  int n = CG * HW; // 64*1024 = 65536
  float s = 0.f, s2 = 0.f;
  for (int i = threadIdx.x; i < n; i += 256) {
    float v = xb[i];
    s += v; s2 += v * v;
  }
  __shared__ float r1[256], r2[256];
  r1[threadIdx.x] = s; r2[threadIdx.x] = s2;
  __syncthreads();
  for (int off = 128; off > 0; off >>= 1) {
    if (threadIdx.x < off) {
      r1[threadIdx.x] += r1[threadIdx.x + off];
      r2[threadIdx.x] += r2[threadIdx.x + off];
    }
    __syncthreads();
  }
  float inv_n = 1.0f / (float)n;
  float mean = r1[0] * inv_n;
  float var  = r2[0] * inv_n - mean * mean;
  float rstd = rsqrtf(var + 1e-5f);
  for (int i = threadIdx.x; i < n; i += 256) {
    int cl = i / HW, p = i - cl * HW;
    int c = c0 + cl;
    float v = (xb[i] - mean) * rstd * gw[c] + gb[c];
    xnT[((size_t)b * HW + p) * C + c] = (_Float16)v;
  }
}

// ---------------------------------------------------------------------------
// Generic WMMA GEMM: D[m][n] = sum_k A[m][k] * Bt[n][k]  (both row-major in K)
// Block tile 128(M) x 128(N), K-step 32, double-buffered LDS, 8 waves,
// each wave a 32x64 patch = 8 WMMA tiles (2 A frags x 4 B frags per K-step).
// ---------------------------------------------------------------------------
template <typename Epi>
__global__ __launch_bounds__(256) void gemm_f16_wmma(const _Float16* __restrict__ A, long sA, int lda,
                                                     const _Float16* __restrict__ Bt, long sB, int ldb,
                                                     int K, Epi epi) {
  __shared__ __align__(16) _Float16 As[2][128][32];
  __shared__ __align__(16) _Float16 Bs[2][128][32];

  const int tid  = threadIdx.x;
  const int b    = blockIdx.z;
  const int wave = tid >> 5;
  const int lane = tid & 31;
  const int wm   = (wave & 3) * 32;   // wave M offset inside block tile
  const int wn   = (wave >> 2) * 64;  // wave N offset inside block tile
  const int lr   = lane & 15;
  const int kh   = lane >> 4;

  const _Float16* Ab = A  + (size_t)b * sA + (size_t)blockIdx.x * 128 * lda;
  const _Float16* Bb = Bt + (size_t)b * sB + (size_t)blockIdx.y * 128 * ldb;

  // tile staging: 128x32 f16 per operand, 16 f16 per thread (two 16B chunks)
  const int r = tid >> 1, h = (tid & 1) * 16;
  const _Float16* Ar = Ab + (size_t)r * lda + h;
  const _Float16* Br = Bb + (size_t)r * ldb + h;

  v8f acc[2][4] = {};

  // prologue: global -> regs -> LDS buffer 0
  v8h sa0 = *(const v8h*)(Ar);
  v8h sa1 = *(const v8h*)(Ar + 8);
  v8h sb0 = *(const v8h*)(Br);
  v8h sb1 = *(const v8h*)(Br + 8);
  *(v8h*)&As[0][r][h] = sa0; *(v8h*)&As[0][r][h + 8] = sa1;
  *(v8h*)&Bs[0][r][h] = sb0; *(v8h*)&Bs[0][r][h + 8] = sb1;

  int cur = 0;
  for (int kk = 0; kk < K; kk += 32) {
    __syncthreads();
    const bool more = (kk + 32) < K;
    if (more) { // prefetch next K-tile into regs; overlaps with WMMA burst below
      sa0 = *(const v8h*)(Ar + kk + 32);
      sa1 = *(const v8h*)(Ar + kk + 40);
      sb0 = *(const v8h*)(Br + kk + 32);
      sb1 = *(const v8h*)(Br + kk + 40);
    }

    v16h af[2], bf[4];
    for (int t = 0; t < 2; ++t) {
      // A fragment 16x32: lane holds row (lane&15), K chunks kh*8..+7 and 16+kh*8..+7
      int ar = wm + t * 16 + lr;
      v8h alo = *(const v8h*)&As[cur][ar][kh * 8];
      v8h ahi = *(const v8h*)&As[cur][ar][16 + kh * 8];
      af[t] = SHUF16(alo, ahi);
    }
    for (int t = 0; t < 4; ++t) {
      // B fragment 32x16: lane holds col (lane&15), K chunks kh*8..+7 and 16+kh*8..+7
      int br = wn + t * 16 + lr;
      v8h blo = *(const v8h*)&Bs[cur][br][kh * 8];
      v8h bhi = *(const v8h*)&Bs[cur][br][16 + kh * 8];
      bf[t] = SHUF16(blo, bhi);
    }

    for (int tm = 0; tm < 2; ++tm)
      for (int tn = 0; tn < 4; ++tn)
        acc[tm][tn] = __builtin_amdgcn_wmma_f32_16x16x32_f16(
            false, af[tm], false, bf[tn], (short)0, acc[tm][tn], false, false);

    if (more) { // fill the other buffer (no barrier needed: disjoint from 'cur')
      int nxt = cur ^ 1;
      *(v8h*)&As[nxt][r][h] = sa0; *(v8h*)&As[nxt][r][h + 8] = sa1;
      *(v8h*)&Bs[nxt][r][h] = sb0; *(v8h*)&Bs[nxt][r][h + 8] = sb1;
    }
    cur ^= 1;
  }

  // C/D layout: lane (lr, kh) VGPR i holds element (M = i + 8*kh, N = lr)
  for (int tm = 0; tm < 2; ++tm)
    for (int tn = 0; tn < 4; ++tn)
      for (int i = 0; i < 8; ++i) {
        int gm = blockIdx.x * 128 + wm + tm * 16 + i + 8 * kh;
        int gn = blockIdx.y * 128 + wn + tn * 16 + lr;
        epi(b, gm, gn, acc[tm][tn][i]);
      }
}

// ---------------------------------------------------------------------------
// Epilogues
// ---------------------------------------------------------------------------
struct EpiQKV { // gm = output channel o (0..1535), gn = pixel p
  const float* __restrict__ bias;
  _Float16 *qT, *kT, *v;
  int C, HW;
  __device__ void operator()(int b, int gm, int gn, float val) const {
    val += bias[gm];
    if (gm < 512)
      qT[((size_t)b * HW + gn) * C + gm] = (_Float16)val;              // q: [p][c]
    else if (gm < 1024)
      kT[((size_t)b * HW + gn) * C + (gm - 512)] = (_Float16)val;      // k: [p][c]
    else
      v[((size_t)b * C + (gm - 1024)) * HW + gn] = (_Float16)val;      // v: [c][p]
  }
};

struct EpiScores { // gm = i, gn = j
  _Float16* S;
  int HW;
  float scale;
  __device__ void operator()(int b, int gm, int gn, float val) const {
    S[((size_t)b * HW + gm) * HW + gn] = (_Float16)(val * scale);
  }
};

struct EpiCtx { // gm = i (pixel), gn = c
  _Float16* ctx;
  int C, HW;
  __device__ void operator()(int b, int gm, int gn, float val) const {
    ctx[((size_t)b * HW + gm) * C + gn] = (_Float16)val;               // ctx: [p][c]
  }
};

struct EpiProj { // gm = o, gn = p ; out = acc + proj_b + residual x
  const float* __restrict__ bias;
  const float* __restrict__ x;
  float* __restrict__ out;
  int C, HW;
  __device__ void operator()(int b, int gm, int gn, float val) const {
    size_t idx = ((size_t)b * C + gm) * HW + gn;
    out[idx] = val + bias[gm] + x[idx];
  }
};

// ---------------------------------------------------------------------------
// K4: row softmax over f16 scores, in place. One block per (b, i) row of 1024.
// ---------------------------------------------------------------------------
__global__ __launch_bounds__(256) void softmax_kernel(_Float16* __restrict__ S, int HW) {
  size_t row = blockIdx.x;
  _Float16* r = S + row * (size_t)HW;
  int t = threadIdx.x;
  float v[4];
  float mx = -1e30f;
  for (int j = 0; j < 4; ++j) {
    v[j] = (float)r[t + 256 * j];
    mx = fmaxf(mx, v[j]);
  }
  __shared__ float red[256];
  red[t] = mx; __syncthreads();
  for (int off = 128; off > 0; off >>= 1) {
    if (t < off) red[t] = fmaxf(red[t], red[t + off]);
    __syncthreads();
  }
  mx = red[0];
  __syncthreads();
  float s = 0.f;
  for (int j = 0; j < 4; ++j) { v[j] = __expf(v[j] - mx); s += v[j]; }
  red[t] = s; __syncthreads();
  for (int off = 128; off > 0; off >>= 1) {
    if (t < off) red[t] += red[t + off];
    __syncthreads();
  }
  float inv = 1.0f / red[0];
  for (int j = 0; j < 4; ++j) r[t + 256 * j] = (_Float16)(v[j] * inv);
}

// ---------------------------------------------------------------------------
// Host orchestration
// ---------------------------------------------------------------------------
extern "C" void kernel_launch(void* const* d_in, const int* in_sizes, int n_in,
                              void* d_out, int out_size, void* d_ws, size_t ws_size,
                              hipStream_t stream) {
  constexpr int B = 16, C = 512, H = 32, W = 32, HW = H * W, G = 8, CG = C / G;

  const float* x      = (const float*)d_in[0];
  const float* norm_w = (const float*)d_in[1];
  const float* norm_b = (const float*)d_in[2];
  const float* qkv_w  = (const float*)d_in[3];
  const float* qkv_b  = (const float*)d_in[4];
  const float* proj_w = (const float*)d_in[5];
  const float* proj_b = (const float*)d_in[6];
  float* out = (float*)d_out;

  // workspace carve-up (256B aligned)
  char* p = (char*)d_ws;
  auto alloc = [&](size_t bytes) -> void* {
    void* r = (void*)p;
    p += (bytes + 255) & ~(size_t)255;
    return r;
  };
  _Float16* wq  = (_Float16*)alloc((size_t)3 * C * C * 2);       // qkv_w f16
  _Float16* wp  = (_Float16*)alloc((size_t)C * C * 2);           // proj_w f16
  _Float16* xnT = (_Float16*)alloc((size_t)B * HW * C * 2);      // xnT, reused as ctx
  _Float16* qT  = (_Float16*)alloc((size_t)B * HW * C * 2);
  _Float16* kT  = (_Float16*)alloc((size_t)B * HW * C * 2);
  _Float16* vB  = (_Float16*)alloc((size_t)B * C * HW * 2);
  _Float16* S   = (_Float16*)alloc((size_t)B * HW * HW * 2);     // scores/probs
  _Float16* ctx = xnT; // xnT fully consumed before ctx is produced

  // K0: weight conversion
  cvt_weights<<<(3 * C * C + 255) / 256, 256, 0, stream>>>(
      qkv_w, proj_w, wq, wp, 3 * C * C, C * C);

  // K1: GroupNorm -> xnT [b][p][c]
  groupnorm_kernel<<<B * G, 256, 0, stream>>>(x, norm_w, norm_b, xnT, C, HW, CG);

  // K2: QKV GEMM  (M=1536, N=1024, K=512): A = wq [o][c], Bt = xnT rows [p][c]
  {
    EpiQKV e{qkv_b, qT, kT, vB, C, HW};
    gemm_f16_wmma<EpiQKV><<<dim3(3 * C / 128, HW / 128, B), 256, 0, stream>>>(
        wq, 0L, C, xnT, (long)HW * C, C, C, e);
  }

  // K3: scores (M=1024, N=1024, K=512): A = qT rows [i][c], Bt = kT rows [j][c]
  {
    EpiScores e{S, HW, 0.04419417382415922f /* 1/sqrt(512) */};
    gemm_f16_wmma<EpiScores><<<dim3(HW / 128, HW / 128, B), 256, 0, stream>>>(
        qT, (long)HW * C, C, kT, (long)HW * C, C, C, e);
  }

  // K4: softmax rows
  softmax_kernel<<<B * HW, 256, 0, stream>>>(S, HW);

  // K5: ctx = P @ v^T (M=1024 i, N=512 c, K=1024 j): A = P [i][j], Bt = v rows [c][j]
  {
    EpiCtx e{ctx, C, HW};
    gemm_f16_wmma<EpiCtx><<<dim3(HW / 128, C / 128, B), 256, 0, stream>>>(
        S, (long)HW * HW, HW, vB, (long)C * HW, HW, HW, e);
  }

  // K6: proj + bias + residual (M=512 o, N=1024 p, K=512 c): A = wp, Bt = ctx rows [p][c]
  {
    EpiProj e{proj_b, x, out, C, HW};
    gemm_f16_wmma<EpiProj><<<dim3(C / 128, HW / 128, B), 256, 0, stream>>>(
        wp, 0L, C, ctx, (long)HW * C, C, C, e);
  }
}
